// Adapter_55104430408051
// MI455X (gfx1250) — matrix operand order repaired and checked by reading, not compiled, not run on hardware
//
#include <hip/hip_runtime.h>

typedef __attribute__((ext_vector_type(2)))  _Float16 h2;
typedef __attribute__((ext_vector_type(4)))  _Float16 h4;
typedef __attribute__((ext_vector_type(8)))  _Float16 h8;
typedef __attribute__((ext_vector_type(16))) _Float16 v16h;
typedef __attribute__((ext_vector_type(8)))  float    v8f;

#define M_ 2
#define N_ 8
#define C_ 1024
#define D_ 256
#define B_ 32
#define S_ 512

#define ROWS 64        // S rows per workgroup
#define KT   32        // K-chunk == one WMMA K per tile
#define NB   256       // output-column chunk in stage 2
#define AST  40        // f16 tile row stride (mult of 8 -> 16B-aligned b128 rows)
#define ZST  264       // z row stride in halfs (256 + 8 pad, 16B-aligned)

__launch_bounds__(256)
__global__ void adapter_fused_kernel(const float* __restrict__ x,
                                     const int* __restrict__ expert_index,
                                     const float* __restrict__ down_w,
                                     const float* __restrict__ down_b,
                                     const float* __restrict__ up_w,
                                     float* __restrict__ out) {
  // 59,392 B total static LDS
  __shared__ __align__(16) _Float16 xa[ROWS * AST];  // A tile (row-major, K contig)
  __shared__ __align__(16) _Float16 wb[256 * AST];   // B tile (transposed: [n][k])
  __shared__ __align__(16) _Float16 zh[ROWS * ZST];  // z block (row-major, K contig)

  const int tid  = threadIdx.x;
  const int wave = tid >> 5;
  const int lane = tid & 31;
  const int kh   = lane >> 4;          // lane half per ISA fragment layouts
  const int l16  = lane & 15;
  const int r0   = (wave >> 1) * 16;   // 4 row tiles: 0,16,32,48
  const int nb0  = (wave & 1) * 128;   // 8 col-tiles of 16 per wave

  const int bid  = blockIdx.x;         // 512 blocks: M * B * (S/ROWS)
  const int s0   = (bid & 7) * ROWS;
  const int b    = (bid >> 3) & 31;
  const int m    = bid >> 8;

  const int e = expert_index[m * B_ + b];
  const float* Xb = x      + (size_t)(b * S_ + s0) * C_;
  const float* Wd = down_w + (size_t)(m * N_ + e) * C_ * D_;
  const float* bd = down_b + (size_t)(m * N_ + e) * D_;
  const float* Wu = up_w   + (size_t)(m * N_ + e) * D_ * C_;
  float*       U  = out    + ((size_t)(m * B_ + b) * S_ + s0) * C_;

  // ---------------- Stage 1: z = silu(X @ Wd + bias) ----------------
  v8f acc[8] = {};

  for (int k0 = 0; k0 < C_; k0 += KT) {
    __syncthreads();
    // X tile [ROWS x 32] f32 -> f16 row-major; b128 global, b64 LDS stores
#pragma unroll
    for (int i = 0; i < (ROWS * KT) / (256 * 4); ++i) {   // 2
      int li4 = (tid + i * 256) * 4;
      int row = li4 >> 5;
      int col = li4 & (KT - 1);
      float4 v = *(const float4*)&Xb[(size_t)row * C_ + k0 + col];
      h4 p = {(_Float16)v.x, (_Float16)v.y, (_Float16)v.z, (_Float16)v.w};
      *(h4*)&xa[row * AST + col] = p;
    }
    // Wd tile [32 x 256] -> f16 transposed wb[n][k]. Lane owns column n=tid,
    // loads K-pair, packs to one b32 store (2-way banked, stride 20 words).
#pragma unroll
    for (int kp = 0; kp < KT / 2; ++kp) {
      float a0 = Wd[(size_t)(k0 + 2 * kp + 0) * D_ + tid];
      float a1 = Wd[(size_t)(k0 + 2 * kp + 1) * D_ + tid];
      h2 p = {(_Float16)a0, (_Float16)a1};
      *(h2*)&wb[tid * AST + 2 * kp] = p;
    }
    __syncthreads();

    // A fragment: per-lane K = kh*8+{0..7} and 16+kh*8+{0..7} (ISA 16-bit A)
    h8 alo = *(const h8*)&xa[(r0 + l16) * AST + kh * 8];
    h8 ahi = *(const h8*)&xa[(r0 + l16) * AST + 16 + kh * 8];
    v16h af = __builtin_shufflevector(alo, ahi, 0, 1, 2, 3, 4, 5, 6, 7,
                                      8, 9, 10, 11, 12, 13, 14, 15);
#pragma unroll
    for (int j = 0; j < 8; ++j) {
      int n = nb0 + j * 16 + l16;
      // B fragment: per-lane col n, K = kh*8+{0..7} and 16+kh*8+{0..7}
      h8 blo = *(const h8*)&wb[n * AST + kh * 8];
      h8 bhi = *(const h8*)&wb[n * AST + 16 + kh * 8];
      v16h bf = __builtin_shufflevector(blo, bhi, 0, 1, 2, 3, 4, 5, 6, 7,
                                        8, 9, 10, 11, 12, 13, 14, 15);
      acc[j] = __builtin_amdgcn_wmma_f32_16x16x32_f16(
          false, af, false, bf, (short)0, acc[j], false, false);
    }
  }

  // bias + silu in f32, park z as f16
#pragma unroll
  for (int j = 0; j < 8; ++j) {
    int n = nb0 + j * 16 + l16;
    float bv = bd[n];
#pragma unroll
    for (int r = 0; r < 8; ++r) {
      float v = acc[j][r] + bv;
      float zz = v / (1.0f + __expf(-v));
      zh[(r0 + r + 8 * kh) * ZST + n] = (_Float16)zz;
    }
  }
  __syncthreads();

  // ---------------- Stage 2: U = z @ Wu ----------------
  for (int cb = 0; cb < C_; cb += NB) {
    v8f acc2[8] = {};
    for (int k0 = 0; k0 < D_; k0 += KT) {
      __syncthreads();
      // Wu tile [32 x 256] -> f16 transposed wb[c][k], K-pair packed b32 stores
#pragma unroll
      for (int kp = 0; kp < KT / 2; ++kp) {
        float a0 = Wu[(size_t)(k0 + 2 * kp + 0) * C_ + cb + tid];
        float a1 = Wu[(size_t)(k0 + 2 * kp + 1) * C_ + cb + tid];
        h2 p = {(_Float16)a0, (_Float16)a1};
        *(h2*)&wb[tid * AST + 2 * kp] = p;
      }
      __syncthreads();

      // A from z (K contiguous per lane), reused across all 8 B tiles
      h8 alo = *(const h8*)&zh[(r0 + l16) * ZST + k0 + kh * 8];
      h8 ahi = *(const h8*)&zh[(r0 + l16) * ZST + k0 + 16 + kh * 8];
      v16h af = __builtin_shufflevector(alo, ahi, 0, 1, 2, 3, 4, 5, 6, 7,
                                        8, 9, 10, 11, 12, 13, 14, 15);
#pragma unroll
      for (int j = 0; j < 8; ++j) {
        int n = nb0 + j * 16 + l16;
        h8 blo = *(const h8*)&wb[n * AST + kh * 8];
        h8 bhi = *(const h8*)&wb[n * AST + 16 + kh * 8];
        v16h bf = __builtin_shufflevector(blo, bhi, 0, 1, 2, 3, 4, 5, 6, 7,
                                          8, 9, 10, 11, 12, 13, 14, 15);
        acc2[j] = __builtin_amdgcn_wmma_f32_16x16x32_f16(
            false, af, false, bf, (short)0, acc2[j], false, false);
      }
    }
    // coalesced f32 stores (64 B per half-wave)
#pragma unroll
    for (int j = 0; j < 8; ++j) {
      int n = cb + nb0 + j * 16 + l16;
#pragma unroll
      for (int r = 0; r < 8; ++r) {
        U[(size_t)(r0 + r + 8 * kh) * C_ + n] = acc2[j][r];
      }
    }
  }
}

extern "C" void kernel_launch(void* const* d_in, const int* in_sizes, int n_in,
                              void* d_out, int out_size, void* d_ws, size_t ws_size,
                              hipStream_t stream) {
  const float* x            = (const float*)d_in[0];
  const int*   expert_index = (const int*)d_in[1];
  const float* down_w       = (const float*)d_in[2];
  const float* down_b       = (const float*)d_in[3];
  const float* up_w         = (const float*)d_in[4];
  float*       out          = (float*)d_out;

  dim3 grid(M_ * B_ * (S_ / ROWS));   // 512 workgroups
  dim3 block(256);                    // 8 wave32
  adapter_fused_kernel<<<grid, block, 0, stream>>>(x, expert_index, down_w,
                                                   down_b, up_w, out);
}
